// LGCN_19670950216172
// MI455X (gfx1250) — matrix operand order repaired and checked by reading, not compiled, not run on hardware
//
#include <hip/hip_runtime.h>
#include <hip/hip_fp16.h>

#define N_NODES 10000
#define N_EDGES 320000
#define T_STEPS 24
#define H 64
#define G4 256   // 4*H gates per node

typedef __attribute__((ext_vector_type(16))) _Float16 v16h;
typedef __attribute__((ext_vector_type(8)))  float    v8f;

__device__ __forceinline__ float sigf(float x) { return 1.0f / (1.0f + __expf(-x)); }

// ---------------- utility kernels ----------------
__global__ void k_zero(float* __restrict__ p, int n) {
  int i = blockIdx.x * blockDim.x + threadIdx.x;
  if (i < n) p[i] = 0.0f;
}

__global__ void k_deg(const int* __restrict__ src, const int* __restrict__ dst,
                      float* __restrict__ dout, float* __restrict__ din, int e) {
  int i = blockIdx.x * blockDim.x + threadIdx.x;
  if (i < e) {
    atomicAdd(&dout[src[i]], 1.0f);
    atomicAdd(&din[dst[i]], 1.0f);
  }
}

__global__ void k_norm(float* __restrict__ d, int n) {
  int i = blockIdx.x * blockDim.x + threadIdx.x;
  if (i < n) {
    float v = d[i];
    d[i] = (v > 0.0f) ? rsqrtf(v) : 0.0f;
  }
}

// Pack 64x64 fp32 W[k][j] into f16 WMMA B-fragment order:
//   frag[kt][jt][lane][e]  with lane = (j%16) + 16*(k%32>=16), e = (k%32)%16
__global__ void k_pack_frag(const float* __restrict__ w, _Float16* __restrict__ o) {
  int idx = blockIdx.x * blockDim.x + threadIdx.x;
  if (idx >= 64 * 64) return;
  int k = idx >> 6, j = idx & 63;
  int kt = k >> 5, kk = k & 31;
  int jt = j >> 4, nn = j & 15;
  int lane = nn + 16 * (kk >> 4);
  int e = kk & 15;
  o[(((kt * 4 + jt) * 32) + lane) * 16 + e] = (_Float16)w[idx];
}

// ---------------- LSTM step (per-node weights: HBM streaming, fp32 VALU) -------------
// block = 256 threads = one node; thread g computes gate g via 64-wide dot with h.
__global__ __launch_bounds__(256) void k_lstm(
    const float* __restrict__ blob, const float* __restrict__ Wih,
    const float* __restrict__ bih,  const float* __restrict__ Whh,
    const float* __restrict__ bhh,  float* __restrict__ h, float* __restrict__ c, int t) {
  __shared__ float hs[H];
  __shared__ float gates[G4];
  const int n = blockIdx.x;
  const int g = threadIdx.x;
  if (g < H) hs[g] = h[(size_t)n * H + g];
  __syncthreads();

  const float x = blob[(size_t)n * T_STEPS + t];
  float acc = Wih[(size_t)n * G4 + g] * x + bih[(size_t)n * G4 + g] + bhh[(size_t)n * G4 + g];

  const float4* wr = (const float4*)(Whh + ((size_t)n * G4 + g) * H);
  __builtin_prefetch(wr + 16, 0, 1);  // global_prefetch_b8: next 256B of the W_hh stream
#pragma unroll
  for (int k4 = 0; k4 < 16; ++k4) {
    float4 w = wr[k4];
    acc += w.x * hs[4 * k4 + 0] + w.y * hs[4 * k4 + 1] +
           w.z * hs[4 * k4 + 2] + w.w * hs[4 * k4 + 3];
  }
  gates[g] = acc;
  __syncthreads();

  if (g < H) {
    float i_ = sigf(gates[g]);
    float f_ = sigf(gates[H + g]);
    float g_ = tanhf(gates[2 * H + g]);
    float o_ = sigf(gates[3 * H + g]);
    float cn = f_ * c[(size_t)n * H + g] + i_ * g_;
    c[(size_t)n * H + g] = cn;
    h[(size_t)n * H + g] = o_ * tanhf(cn);
  }
}

// ---------------- edge scatter: agg[dst] += feat[src] * norm_src[src] ----------------
__global__ __launch_bounds__(256) void k_msg(
    const float* __restrict__ feat, const float* __restrict__ nsrc,
    const int* __restrict__ src, const int* __restrict__ dst, float* __restrict__ agg) {
  const int wid = threadIdx.x >> 5;
  const int lane = threadIdx.x & 31;
  const int e = blockIdx.x * 8 + wid;
  if (e >= N_EDGES) return;
  const int s = src[e], d = dst[e];
  const float sc = nsrc[s];
  const float* fr = feat + (size_t)s * H;
  float* ar = agg + (size_t)d * H;
  atomicAdd(&ar[lane],      fr[lane]      * sc);
  atomicAdd(&ar[lane + 32], fr[lane + 32] * sc);
}

// ---------------- WMMA GEMM: out = relu((agg * norm_dst) @ W + b) --------------------
// block = 128 threads (4 waves) covering 64 nodes; each wave owns a 16x64 slab.
// A staged in LDS in fragment order (2x ds_load_b128 per fragment); B read from
// prepacked global fragments (L2-hot). 8x v_wmma_f32_16x16x32_f16 per wave.
__global__ __launch_bounds__(128) void k_gemm_relu(
    const float* __restrict__ agg, const float* __restrict__ ndst,
    const _Float16* __restrict__ Wfrag, const float* __restrict__ bias,
    float* __restrict__ out) {
  __shared__ __align__(32) _Float16 Af[4][2][32][16];  // [mtile][kt][lane][elem] = 8KB

  const int tid  = threadIdx.x;
  const int wave = tid >> 5;
  const int lane = tid & 31;
  const int n0 = blockIdx.x * 64;

  // stage A (scaled, f16) directly in contiguous-k fragment order (k = 16*(lane/16) + e), same as B packing
  {
    const int r  = tid >> 1;           // local row 0..63
    const int c0 = (tid & 1) * 32;     // k chunk
    const int node = n0 + r;
    const bool ok = (node < N_NODES);
    const float sc = ok ? ndst[node] : 0.0f;
    const float* ap = agg + (size_t)node * H;
    const int mt = r >> 4, m = r & 15;
#pragma unroll
    for (int j = 0; j < 32; ++j) {
      const int k = c0 + j;
      const int kt = k >> 5, kk = k & 31;
      const int hi = kk >> 4;
      const int e = kk & 15;
      float val = ok ? ap[k] * sc : 0.0f;
      Af[mt][kt][m + 16 * hi][e] = (_Float16)val;
    }
  }
  __syncthreads();

  // N_NODES % 16 == 0 -> each 16-row tile fully in or fully out: wave-uniform guard
  if (n0 + 16 * wave >= N_NODES) return;

  const v16h a0 = *(const v16h*)&Af[wave][0][lane][0];
  const v16h a1 = *(const v16h*)&Af[wave][1][lane][0];
  const v16h* Bf = (const v16h*)Wfrag;          // [kt][jt][lane]

  const int nn = lane & 15;
  const int mbase = (lane >> 4) * 8;            // D: VGPR v <-> M = v + 8*(lane>>4)
#pragma unroll
  for (int jt = 0; jt < 4; ++jt) {
    v16h b0 = Bf[(0 * 4 + jt) * 32 + lane];
    v16h b1 = Bf[(1 * 4 + jt) * 32 + lane];
    v8f acc = {};
    acc = __builtin_amdgcn_wmma_f32_16x16x32_f16(false, a0, false, b0,
                                                 (short)0, acc, false, false);
    acc = __builtin_amdgcn_wmma_f32_16x16x32_f16(false, a1, false, b1,
                                                 (short)0, acc, false, false);
    const int col = 16 * jt + nn;
    const float bcol = bias[col];
    const size_t base = (size_t)(n0 + 16 * wave + mbase) * H + col;
#pragma unroll
    for (int v = 0; v < 8; ++v)
      out[base + (size_t)v * H] = fmaxf(acc[v] + bcol, 0.0f);
  }
}

// ---------------- final 64->1 projection: y[n,t] = dot(agg[n]*norm, Wg2) + bg2 -------
__global__ __launch_bounds__(256) void k_proj(
    const float* __restrict__ agg, const float* __restrict__ ndst,
    const float* __restrict__ Wg2, const float* __restrict__ bg2,
    float* __restrict__ out, int t) {
  const int wid = threadIdx.x >> 5;
  const int lane = threadIdx.x & 31;
  const int n = blockIdx.x * 8 + wid;
  if (n >= N_NODES) return;
  const float sc = ndst[n];
  const float* ap = agg + (size_t)n * H;
  float v = ap[lane] * Wg2[lane] + ap[lane + 32] * Wg2[lane + 32];
  v *= sc;
  for (int o = 16; o > 0; o >>= 1) v += __shfl_down(v, o, 32);
  if (lane == 0) out[(size_t)n * T_STEPS + t] = v + bg2[0];
}

// ---------------- driver ----------------
extern "C" void kernel_launch(void* const* d_in, const int* in_sizes, int n_in,
                              void* d_out, int out_size, void* d_ws, size_t ws_size,
                              hipStream_t stream) {
  const float* blob = (const float*)d_in[0];
  const float* Wih  = (const float*)d_in[1];   // (N, 256, 1) -> Wx
  const float* bih  = (const float*)d_in[2];
  const float* Whh  = (const float*)d_in[3];
  const float* bhh  = (const float*)d_in[4];
  const float* Wg0  = (const float*)d_in[5];
  const float* bg0  = (const float*)d_in[6];
  const float* Wg1  = (const float*)d_in[7];
  const float* bg1  = (const float*)d_in[8];
  const float* Wg2  = (const float*)d_in[9];
  const float* bg2  = (const float*)d_in[10];
  const int* esrc   = (const int*)d_in[11];
  const int* edst   = (const int*)d_in[12];
  float* out = (float*)d_out;

  // ws layout: prepacked f16 weight fragments first (32B-aligned base), then fp32 state
  _Float16* W0h = (_Float16*)d_ws;             // 4096 halves
  _Float16* W1h = W0h + 64 * 64;               // 4096 halves
  float* h    = (float*)(W1h + 64 * 64);       // N*H
  float* c    = h + (size_t)N_NODES * H;
  float* agg  = c + (size_t)N_NODES * H;
  float* z    = agg + (size_t)N_NODES * H;
  float* nsrc = z + (size_t)N_NODES * H;
  float* ndst = nsrc + N_NODES;

  const int NF = N_NODES * H;

  // init: h,c = 0; norms = rsqrt(degree); weights -> f16 fragments
  k_zero<<<(2 * NF + 255) / 256, 256, 0, stream>>>(h, 2 * NF);
  k_zero<<<(2 * N_NODES + 255) / 256, 256, 0, stream>>>(nsrc, 2 * N_NODES);
  k_deg<<<(N_EDGES + 255) / 256, 256, 0, stream>>>(esrc, edst, nsrc, ndst, N_EDGES);
  k_norm<<<(2 * N_NODES + 255) / 256, 256, 0, stream>>>(nsrc, 2 * N_NODES);
  k_pack_frag<<<16, 256, 0, stream>>>(Wg0, W0h);
  k_pack_frag<<<16, 256, 0, stream>>>(Wg1, W1h);

  const int gemm_blocks = (N_NODES + 63) / 64;  // 157

  for (int t = 0; t < T_STEPS; ++t) {
    k_lstm<<<N_NODES, 256, 0, stream>>>(blob, Wih, bih, Whh, bhh, h, c, t);

    // gconv layer 0: h -> z
    k_zero<<<(NF + 255) / 256, 256, 0, stream>>>(agg, NF);
    k_msg<<<N_EDGES / 8, 256, 0, stream>>>(h, nsrc, esrc, edst, agg);
    k_gemm_relu<<<gemm_blocks, 128, 0, stream>>>(agg, ndst, W0h, bg0, z);

    // gconv layer 1: z -> z
    k_zero<<<(NF + 255) / 256, 256, 0, stream>>>(agg, NF);
    k_msg<<<N_EDGES / 8, 256, 0, stream>>>(z, nsrc, esrc, edst, agg);
    k_gemm_relu<<<gemm_blocks, 128, 0, stream>>>(agg, ndst, W1h, bg1, z);

    // gconv layer 2: z -> out[:, t]
    k_zero<<<(NF + 255) / 256, 256, 0, stream>>>(agg, NF);
    k_msg<<<N_EDGES / 8, 256, 0, stream>>>(z, nsrc, esrc, edst, agg);
    k_proj<<<(N_NODES + 7) / 8, 256, 0, stream>>>(agg, ndst, Wg2, bg2, out, t);
  }
}
